// KMeans_35991825940871
// MI455X (gfx1250) — matrix operand order repaired and checked by reading, not compiled, not run on hardware
//
#include <hip/hip_runtime.h>
#include <stdint.h>

#define N_PTS   262144
#define DIM     128
#define K_MEANS 1024

typedef __attribute__((ext_vector_type(16))) __bf16 v16bf;
typedef __attribute__((ext_vector_type(8)))  float  v8f;

union Frag32B { uint4 u[2]; v16bf v; };

__device__ __forceinline__ unsigned short f32_to_bf16_rne(float x) {
    unsigned u = __float_as_uint(x);
    unsigned r = (u + 0x7FFFu + ((u >> 16) & 1u)) >> 16;   // round-to-nearest-even
    return (unsigned short)r;
}

// One wave (32 lanes) per row of a [nrows x 128] f32 matrix:
// writes bf16 row + fp32 squared norm.
__global__ __launch_bounds__(256)
void convert_rows_kernel(const float* __restrict__ src,
                         unsigned short* __restrict__ dst,
                         float* __restrict__ norms, int nrows) {
    const int wave = (blockIdx.x * blockDim.x + threadIdx.x) >> 5;
    const int lane = threadIdx.x & 31;
    if (wave >= nrows) return;
    const float4 f = *reinterpret_cast<const float4*>(src + (size_t)wave * DIM + lane * 4);
    ushort4 p;
    p.x = f32_to_bf16_rne(f.x);
    p.y = f32_to_bf16_rne(f.y);
    p.z = f32_to_bf16_rne(f.z);
    p.w = f32_to_bf16_rne(f.w);
    *reinterpret_cast<ushort4*>(dst + (size_t)wave * DIM + lane * 4) = p;
    float s = f.x * f.x + f.y * f.y + f.z * f.z + f.w * f.w;
#pragma unroll
    for (int off = 16; off >= 1; off >>= 1) s += __shfl_xor(s, off, 32);
    if (lane == 0) norms[wave] = s;
}

#define TILE_MEANS 64
#define LDS_STRIDE 136   // 128 bf16 + 8 pad (16B) -> conflict-free ds_load_b128

// Each wave owns a 32-row X strip (two register-resident 16x128 A tiles), so
// every B fragment read from LDS feeds TWO WMMAs (1 ds_load_b128 per WMMA).
// Means tiles are double-buffered in LDS via async global->LDS DMA.
__global__ __launch_bounds__(256)
void kmeans_min_kernel(const unsigned short* __restrict__ Xb,
                       const unsigned short* __restrict__ Mb,
                       const float* __restrict__ x2,
                       const float* __restrict__ m2,
                       float* __restrict__ out) {
    __shared__ unsigned short smem[2][TILE_MEANS * LDS_STRIDE];

    const int tid  = threadIdx.x;
    const int lane = tid & 31;
    const int wave = tid >> 5;
    const int hf   = lane >> 4;     // half-wave: 0 or 1
    const int l16  = lane & 15;

    const int row_base = blockIdx.x * 256 + wave * 32;

    // ---- A fragments: 2 x (16 rows x 128 K) in registers, 4 k-steps of 32 ----
    // lanes 0-15: K = d0+0..7 , d0+16..23 ; lanes 16-31: K = d0+8..15 , d0+24..31
    Frag32B a0[4], a1[4];
    {
        const unsigned short* xr0 = Xb + (size_t)(row_base + l16) * DIM;
        const unsigned short* xr1 = Xb + (size_t)(row_base + 16 + l16) * DIM;
#pragma unroll
        for (int s = 0; s < 4; ++s) {
            const int d0 = s * 32 + hf * 8;
            a0[s].u[0] = *reinterpret_cast<const uint4*>(xr0 + d0);
            a0[s].u[1] = *reinterpret_cast<const uint4*>(xr0 + d0 + 16);
            a1[s].u[0] = *reinterpret_cast<const uint4*>(xr1 + d0);
            a1[s].u[1] = *reinterpret_cast<const uint4*>(xr1 + d0 + 16);
        }
    }

    float minv0[8], minv1[8];
#pragma unroll
    for (int r = 0; r < 8; ++r) { minv0[r] = 3.4028235e38f; minv1[r] = 3.4028235e38f; }

    // Async DMA staging: 64 means x 128 K bf16 tile (16 KB) -> LDS, 4 x b128/lane.
    // GVS mode: mem = SADDR(64b SGPR) + VADDR(32b VGPR offset); dst = LDS byte addr.
    auto stage = [&](int tile, int buf) {
        const unsigned src_base = (unsigned)((size_t)tile * TILE_MEANS * DIM * 2u);
#pragma unroll
        for (int i = 0; i < 4; ++i) {
            const int idx = i * 256 + tid;   // 1024 chunks of 8 bf16 (16B)
            const int m   = idx >> 4;        // mean within tile
            const int p   = idx & 15;        // 16B chunk within mean row
            const unsigned voff   = src_base + (unsigned)(m * DIM + p * 8) * 2u;
            const unsigned ldsoff =
                (unsigned)(uintptr_t)(&smem[buf][m * LDS_STRIDE + p * 8]);
            asm volatile("global_load_async_to_lds_b128 %0, %1, %2"
                         :
                         : "v"(ldsoff), "v"(voff), "s"(Mb)
                         : "memory");
        }
    };

    stage(0, 0);

    const int NTILES = K_MEANS / TILE_MEANS;   // 16
    for (int t = 0; t < NTILES; ++t) {
        // my async copies for buffer t landed; barrier publishes all waves'
        asm volatile("s_wait_asynccnt 0x0" ::: "memory");
        __syncthreads();
        if (t + 1 < NTILES) stage(t + 1, (t + 1) & 1);
        const unsigned short* buf = smem[t & 1];

#pragma unroll
        for (int sub = 0; sub < 4; ++sub) {
            const int mcol = sub * 16 + l16;   // mean (B column) this lane holds
            v8f c0 = {}, c1 = {};
#pragma unroll
            for (int s = 0; s < 4; ++s) {
                // B fragment: lanes 0-15 hold K=d0+0..7, d0+16..23; lanes 16-31 K=d0+8..15, d0+24..31
                Frag32B b;
                const unsigned short* bp = buf + mcol * LDS_STRIDE + s * 32 + hf * 8;
                b.u[0] = *reinterpret_cast<const uint4*>(bp);
                b.u[1] = *reinterpret_cast<const uint4*>(bp + 16);
                c0 = __builtin_amdgcn_wmma_f32_16x16x32_bf16(
                        false, a0[s].v, false, b.v, (short)0, c0, false, false);
                c1 = __builtin_amdgcn_wmma_f32_16x16x32_bf16(
                        false, a1[s].v, false, b.v, (short)0, c1, false, false);
            }
            const int   colg = t * TILE_MEANS + sub * 16 + l16;
            const float m2v  = m2[colg];
#pragma unroll
            for (int r = 0; r < 8; ++r) {
                minv0[r] = fminf(minv0[r], m2v - 2.0f * c0[r]);  // ||m||^2 - 2 x.m
                minv1[r] = fminf(minv1[r], m2v - 2.0f * c1[r]);
            }
        }
    }

    // ---- min over the 16 lanes covering each row-half ----
#pragma unroll
    for (int r = 0; r < 8; ++r) {
#pragma unroll
        for (int off = 8; off >= 1; off >>= 1) {
            minv0[r] = fminf(minv0[r], __shfl_xor(minv0[r], off, 16));
            minv1[r] = fminf(minv1[r], __shfl_xor(minv1[r], off, 16));
        }
    }
    if (l16 == 0) {   // lanes 0 and 16: rows rb+0..7 / rb+8..15 of each A tile
        const int rb0 = row_base + hf * 8;
        const int rb1 = row_base + 16 + hf * 8;
        float s = 0.0f;
#pragma unroll
        for (int r = 0; r < 8; ++r) {
            s += x2[rb0 + r] + minv0[r];
            s += x2[rb1 + r] + minv1[r];
        }
        atomicAdd(out, s);
    }
}

extern "C" void kernel_launch(void* const* d_in, const int* in_sizes, int n_in,
                              void* d_out, int out_size, void* d_ws, size_t ws_size,
                              hipStream_t stream) {
    const float* X     = (const float*)d_in[0];
    const float* means = (const float*)d_in[1];
    float*       out   = (float*)d_out;

    // workspace layout (all 256B aligned)
    char* ws = (char*)d_ws;
    const size_t xb_bytes = (size_t)N_PTS * DIM * 2;        // 67,108,864
    const size_t mb_bytes = (size_t)K_MEANS * DIM * 2;      //    262,144
    const size_t x2_bytes = (size_t)N_PTS * 4;              //  1,048,576
    unsigned short* Xb = (unsigned short*)ws;
    unsigned short* Mb = (unsigned short*)(ws + xb_bytes);
    float*          x2 = (float*)(ws + xb_bytes + mb_bytes);
    float*          m2 = (float*)(ws + xb_bytes + mb_bytes + x2_bytes);

    hipMemsetAsync(out, 0, sizeof(float), stream);

    // bf16 conversion + squared norms (wave-per-row)
    convert_rows_kernel<<<N_PTS / 8, 256, 0, stream>>>(X, Xb, x2, N_PTS);
    convert_rows_kernel<<<K_MEANS / 8, 256, 0, stream>>>(means, Mb, m2, K_MEANS);

    // fused GEMM + row-min + loss reduction
    kmeans_min_kernel<<<N_PTS / 256, 256, 0, stream>>>(Xb, Mb, x2, m2, out);
}
